// Attention_14190571946547
// MI455X (gfx1250) — matrix operand order repaired and checked by reading, not compiled, not run on hardware
//
#include <hip/hip_runtime.h>

// ---------------------------------------------------------------------------
// Attention (B=4, S=2048, D=1024, single head, scale = 1/sqrt(64) = 0.125)
// 4-stage pipeline, all matmuls via v_wmma_f32_16x16x32_bf16 (wave32).
// Wave tile 32x32 (4 WMMAs / k-step), block tile 128x64 (8 waves: 4M x 2N).
// Workspace layout:
//   [0, 48MB)   : Q,K,V as bf16, each [8192][1024]
//   [48MB,112MB): scores f32 [4][2048][2048]; after softmax the same rows are
//                 overwritten in-place with bf16 P (row stride stays 8192 B).
// ---------------------------------------------------------------------------

typedef __attribute__((ext_vector_type(16))) __bf16 v16bf;
typedef __attribute__((ext_vector_type(8)))  float  v8f;

#define B_ 4
#define S_ 2048
#define D_ 1024
#define SCALE_ 0.125f

__device__ __forceinline__ __bf16 f2bf(float x) { return (__bf16)x; }

__device__ __forceinline__ v8f wmma_bf16(v16bf a, v16bf b, v8f c) {
  return __builtin_amdgcn_wmma_f32_16x16x32_bf16(
      /*neg_a=*/false, a, /*neg_b=*/false, b,
      /*c_mod=*/(short)0, c, /*reuse_a=*/false, /*reuse_b=*/false);
}

union frag_u { v16bf v; uint4 q[2]; };

// A fragment (16x32, MxK) from row-major bf16 matrix, leading dim ld.
// Lanes 0-15: row = lane, K = 0..15; lanes 16-31: K = 16..31 (contiguous).
__device__ __forceinline__ v16bf load_a_bf16(const __bf16* A, int row0, int k0,
                                             int ld, int lane) {
  const __bf16* r = A + (size_t)(row0 + (lane & 15)) * ld + k0;
  const int kh = (lane >> 4) << 4;
  frag_u u;
  u.q[0] = *(const uint4*)(r + kh);        // 8 bf16, 16B aligned
  u.q[1] = *(const uint4*)(r + kh + 8);
  return u.v;
}

// Same, but f32 source with on-the-fly native bf16 conversion.
__device__ __forceinline__ v16bf load_a_f32(const float* A, int row0, int k0,
                                            int ld, int lane) {
  const float* r = A + (size_t)(row0 + (lane & 15)) * ld + k0;
  const int kh = (lane >> 4) << 4;
  float4 x0 = ((const float4*)(r + kh))[0];
  float4 x1 = ((const float4*)(r + kh))[1];
  float4 y0 = ((const float4*)(r + kh + 8))[0];
  float4 y1 = ((const float4*)(r + kh + 8))[1];
  v16bf a;
  a[0]  = f2bf(x0.x); a[1]  = f2bf(x0.y); a[2]  = f2bf(x0.z); a[3]  = f2bf(x0.w);
  a[4]  = f2bf(x1.x); a[5]  = f2bf(x1.y); a[6]  = f2bf(x1.z); a[7]  = f2bf(x1.w);
  a[8]  = f2bf(y0.x); a[9]  = f2bf(y0.y); a[10] = f2bf(y0.z); a[11] = f2bf(y0.w);
  a[12] = f2bf(y1.x); a[13] = f2bf(y1.y); a[14] = f2bf(y1.z); a[15] = f2bf(y1.w);
  return a;
}

// B fragment (32x16, KxN) where column n is contiguous in memory
// (source row-major with rows indexed by n, e.g. K-matrix for Q*K^T).
// Lane layout: n = col0 + (lane&15); K = 16*(lane>=16) + 0..15 (contiguous).
__device__ __forceinline__ v16bf load_bT_global(const __bf16* Bm, int col0,
                                                int k0, int ld, int lane) {
  const __bf16* r =
      Bm + (size_t)(col0 + (lane & 15)) * ld + k0 + ((lane >> 4) << 4);
  frag_u u;
  u.q[0] = ((const uint4*)r)[0];
  u.q[1] = ((const uint4*)r)[1];
  return u.v;
}

// B fragment from LDS-transposed tile BT[n][k] (row 16B-aligned).
__device__ __forceinline__ v16bf load_bT_lds(const __bf16* row, int kk) {
  frag_u u;
  u.q[0] = *(const uint4*)(row + kk);
  u.q[1] = *(const uint4*)(row + kk + 8);
  return u.v;
}

// ---------------------------------------------------------------------------
// Kernel 1: Q/K/V projection.  C[8192x1024] = X[8192x1024](f32) @ W[1024x1024]
// z = 0/1/2 selects the weight slice; output written as bf16.
// W tile transposed through LDS so B fragments read contiguous bf16.
// ---------------------------------------------------------------------------
__global__ __launch_bounds__(256) void qkv_proj_kernel(
    const float* __restrict__ X, const float* __restrict__ Wall,
    __bf16* __restrict__ QKV) {
  __shared__ __bf16 BT[64][40];  // [n][k] transposed tile; rows 16B-aligned

  const int tid  = threadIdx.x;
  const int lane = tid & 31, wave = tid >> 5;
  const int wm = wave & 3, wn = wave >> 2;
  const int col0 = blockIdx.x * 64;
  const int row0 = blockIdx.y * 128;
  const int z    = blockIdx.z;
  const float* W  = Wall + (size_t)z * D_ * D_;
  __bf16*     Out = QKV  + (size_t)z * (B_ * S_) * D_;

  v8f acc00 = {}, acc01 = {}, acc10 = {}, acc11 = {};
  const int nl = tid & 63;
  const int kq = (tid >> 6) << 3;

  for (int kb = 0; kb < D_; kb += 32) {
    // Stage W[kb..kb+31][col0..col0+63] transposed -> BT[n][k] (bf16)
    __bf16 tmp[8];
#pragma unroll
    for (int j = 0; j < 8; ++j)
      tmp[j] = f2bf(W[(size_t)(kb + kq + j) * D_ + col0 + nl]);
    *(uint4*)&BT[nl][kq] = *(const uint4*)tmp;  // one ds_store_b128
    __syncthreads();

    v16bf a0 = load_a_f32(X, row0 + wm * 32,      kb, D_, lane);
    v16bf a1 = load_a_f32(X, row0 + wm * 32 + 16, kb, D_, lane);
    const int kk = (lane >> 4) << 4;
    v16bf b0 = load_bT_lds(&BT[wn * 32 + (lane & 15)][0],      kk);
    v16bf b1 = load_bT_lds(&BT[wn * 32 + 16 + (lane & 15)][0], kk);

    acc00 = wmma_bf16(a0, b0, acc00);
    acc01 = wmma_bf16(a0, b1, acc01);
    acc10 = wmma_bf16(a1, b0, acc10);
    acc11 = wmma_bf16(a1, b1, acc11);
    __syncthreads();
  }

  const int ncol = col0 + wn * 32 + (lane & 15);
  const int mr0  = row0 + wm * 32 + ((lane >> 4) << 3);
#pragma unroll
  for (int r = 0; r < 8; ++r) {
    Out[(size_t)(mr0 + r) * D_ + ncol]           = f2bf(acc00[r]);
    Out[(size_t)(mr0 + r) * D_ + ncol + 16]      = f2bf(acc01[r]);
    Out[(size_t)(mr0 + 16 + r) * D_ + ncol]      = f2bf(acc10[r]);
    Out[(size_t)(mr0 + 16 + r) * D_ + ncol + 16] = f2bf(acc11[r]);
  }
}

// ---------------------------------------------------------------------------
// Kernel 2: scores S[b] = (Q[b] @ K[b]^T) * SCALE, f32 output.
// K^T columns == K rows, so B fragments load contiguously from global.
// ---------------------------------------------------------------------------
__global__ __launch_bounds__(256) void scores_kernel(
    const __bf16* __restrict__ QKV, float* __restrict__ Sc) {
  const int tid  = threadIdx.x;
  const int lane = tid & 31, wave = tid >> 5;
  const int wm = wave & 3, wn = wave >> 2;
  const int b  = blockIdx.z;
  const int q0 = blockIdx.y * 128 + wm * 32;
  const int n0 = blockIdx.x * 64 + wn * 32;

  const __bf16* Q = QKV + (size_t)b * S_ * D_;
  const __bf16* K = QKV + (size_t)(B_ * S_) * D_ + (size_t)b * S_ * D_;
  float* Sb = Sc + (size_t)b * S_ * S_;

  v8f acc00 = {}, acc01 = {}, acc10 = {}, acc11 = {};
  for (int kb = 0; kb < D_; kb += 32) {
    if (kb + 32 < D_) {  // global_prefetch_b8 of next k-chunk
      __builtin_prefetch(Q + (size_t)(q0 + (lane & 15)) * D_ + kb + 32, 0, 3);
      __builtin_prefetch(K + (size_t)(n0 + (lane & 15)) * D_ + kb + 32, 0, 3);
    }
    v16bf a0 = load_a_bf16(Q, q0,      kb, D_, lane);
    v16bf a1 = load_a_bf16(Q, q0 + 16, kb, D_, lane);
    v16bf b0 = load_bT_global(K, n0,      kb, D_, lane);
    v16bf b1 = load_bT_global(K, n0 + 16, kb, D_, lane);
    acc00 = wmma_bf16(a0, b0, acc00);
    acc01 = wmma_bf16(a0, b1, acc01);
    acc10 = wmma_bf16(a1, b0, acc10);
    acc11 = wmma_bf16(a1, b1, acc11);
  }

  const int ncol = n0 + (lane & 15);
  const int mr0  = q0 + ((lane >> 4) << 3);
#pragma unroll
  for (int r = 0; r < 8; ++r) {
    Sb[(size_t)(mr0 + r) * S_ + ncol]           = acc00[r] * SCALE_;
    Sb[(size_t)(mr0 + r) * S_ + ncol + 16]      = acc01[r] * SCALE_;
    Sb[(size_t)(mr0 + 16 + r) * S_ + ncol]      = acc10[r] * SCALE_;
    Sb[(size_t)(mr0 + 16 + r) * S_ + ncol + 16] = acc11[r] * SCALE_;
  }
}

// ---------------------------------------------------------------------------
// Kernel 3: row softmax over 2048 f32 scores; write bf16 P in-place over the
// start of the same row (row byte stride unchanged: 8192 B -> P ld = 4096).
// All global reads complete before the first barrier; writes come after.
// ---------------------------------------------------------------------------
__global__ __launch_bounds__(256) void softmax_kernel(float* __restrict__ Sc) {
  const int row = blockIdx.x;  // 0 .. B*S-1
  float* r = Sc + (size_t)row * S_;
  const int tid = threadIdx.x, lane = tid & 31, wave = tid >> 5;
  __shared__ float red[8];

  float v[8];
#pragma unroll
  for (int j = 0; j < 8; ++j) v[j] = r[tid + 256 * j];

  float m = v[0];
#pragma unroll
  for (int j = 1; j < 8; ++j) m = fmaxf(m, v[j]);
#pragma unroll
  for (int o = 16; o > 0; o >>= 1) m = fmaxf(m, __shfl_xor(m, o, 32));
  if (lane == 0) red[wave] = m;
  __syncthreads();
  float mm = red[0];
#pragma unroll
  for (int j = 1; j < 8; ++j) mm = fmaxf(mm, red[j]);

  float s = 0.f;
#pragma unroll
  for (int j = 0; j < 8; ++j) {
    v[j] = __expf(v[j] - mm);
    s += v[j];
  }
#pragma unroll
  for (int o = 16; o > 0; o >>= 1) s += __shfl_xor(s, o, 32);
  __syncthreads();
  if (lane == 0) red[wave] = s;
  __syncthreads();
  float ss = 0.f;
#pragma unroll
  for (int j = 0; j < 8; ++j) ss += red[j];
  const float inv = 1.0f / ss;

  __bf16* p = (__bf16*)r;  // in-place bf16 row, safe: all reads done
#pragma unroll
  for (int j = 0; j < 8; ++j) p[tid + 256 * j] = f2bf(v[j] * inv);
}

// ---------------------------------------------------------------------------
// Kernel 4: O[b] = P[b] (2048x2048 bf16, ld=4096) @ V[b] (2048x1024 bf16).
// V tile transposed through LDS (like kernel 1); f32 output to d_out.
// ---------------------------------------------------------------------------
__global__ __launch_bounds__(256) void av_kernel(
    const __bf16* __restrict__ QKV, const float* __restrict__ Sc,
    float* __restrict__ Out) {
  __shared__ __bf16 BT[64][40];

  const int tid  = threadIdx.x;
  const int lane = tid & 31, wave = tid >> 5;
  const int wm = wave & 3, wn = wave >> 2;
  const int b  = blockIdx.z;
  const int o0 = blockIdx.x * 64;   // output column tile
  const int q0 = blockIdx.y * 128;  // query row tile

  const __bf16* V = QKV + (size_t)2 * (B_ * S_) * D_ + (size_t)b * S_ * D_;
  const __bf16* P = (const __bf16*)(Sc + (size_t)b * S_ * S_);  // ld = 2*S_
  float* O = Out + (size_t)b * S_ * D_;

  v8f acc00 = {}, acc01 = {}, acc10 = {}, acc11 = {};
  const int nl = tid & 63;
  const int kq = (tid >> 6) << 3;
  const int qa = q0 + wm * 32;

  for (int kb = 0; kb < S_; kb += 32) {
    __bf16 tmp[8];
#pragma unroll
    for (int j = 0; j < 8; ++j)
      tmp[j] = V[(size_t)(kb + kq + j) * D_ + o0 + nl];
    *(uint4*)&BT[nl][kq] = *(const uint4*)tmp;  // one ds_store_b128
    __syncthreads();

    if (kb + 32 < S_) {
      __builtin_prefetch(P + (size_t)(qa + (lane & 15)) * (2 * S_) + kb + 32,
                         0, 3);
    }
    v16bf a0 = load_a_bf16(P, qa,      kb, 2 * S_, lane);
    v16bf a1 = load_a_bf16(P, qa + 16, kb, 2 * S_, lane);
    const int kk = (lane >> 4) << 4;
    v16bf b0 = load_bT_lds(&BT[wn * 32 + (lane & 15)][0],      kk);
    v16bf b1 = load_bT_lds(&BT[wn * 32 + 16 + (lane & 15)][0], kk);

    acc00 = wmma_bf16(a0, b0, acc00);
    acc01 = wmma_bf16(a0, b1, acc01);
    acc10 = wmma_bf16(a1, b0, acc10);
    acc11 = wmma_bf16(a1, b1, acc11);
    __syncthreads();
  }

  const int ncol = o0 + wn * 32 + (lane & 15);
  const int mr0  = qa + ((lane >> 4) << 3);
#pragma unroll
  for (int r = 0; r < 8; ++r) {
    O[(size_t)(mr0 + r) * D_ + ncol]           = acc00[r];
    O[(size_t)(mr0 + r) * D_ + ncol + 16]      = acc01[r];
    O[(size_t)(mr0 + 16 + r) * D_ + ncol]      = acc10[r];
    O[(size_t)(mr0 + 16 + r) * D_ + ncol + 16] = acc11[r];
  }
}

// ---------------------------------------------------------------------------
extern "C" void kernel_launch(void* const* d_in, const int* in_sizes, int n_in,
                              void* d_out, int out_size, void* d_ws,
                              size_t ws_size, hipStream_t stream) {
  (void)in_sizes; (void)n_in; (void)out_size; (void)ws_size;
  const float* x = (const float*)d_in[0];   // [4,2048,1024]
  const float* w = (const float*)d_in[1];   // [3,1024,1024]
  float* out = (float*)d_out;               // [4,2048,1024]

  __bf16* qkv = (__bf16*)d_ws;                                   // 48 MB
  float* scores =
      (float*)((char*)d_ws + (size_t)3 * B_ * S_ * D_ * sizeof(__bf16));  // 64 MB

  const dim3 blk(256);
  qkv_proj_kernel<<<dim3(D_ / 64, (B_ * S_) / 128, 3), blk, 0, stream>>>(x, w, qkv);
  scores_kernel<<<dim3(S_ / 64, S_ / 128, B_), blk, 0, stream>>>(qkv, scores);
  softmax_kernel<<<dim3(B_ * S_), blk, 0, stream>>>(scores);
  av_kernel<<<dim3(D_ / 64, S_ / 128, B_), blk, 0, stream>>>(qkv, scores, out);
}
